// RingDilatedAttentionHybridFixed_42614665511002
// MI455X (gfx1250) — matrix operand order repaired and checked by reading, not compiled, not run on hardware
//
#include <hip/hip_runtime.h>

typedef __attribute__((ext_vector_type(16))) __bf16 v16bf;
typedef __attribute__((ext_vector_type(8)))  __bf16 v8bf;
typedef __attribute__((ext_vector_type(4)))  __bf16 v4bf;
typedef __attribute__((ext_vector_type(2)))  __bf16 v2bf;
typedef __attribute__((ext_vector_type(8)))  float  v8f;

#define HD 128                 // head dim
#define NH 16                  // num heads
#define ROWSTRIDE (NH * HD)    // 2048 floats per token
#define SEQ 2048               // dilated sequence length per problem (both groups)
#define BR 128                 // query rows per workgroup (8 waves x 16)
#define BC 32                  // key/value tile
#define NT (SEQ / BC)          // kv iterations
#define KS 136                 // ldsK row stride (bf16 elems), 272B, 16B aligned
#define VS 40                  // ldsV row stride, 80B
#define QS 136                 // ldsQ row stride
#define PS 40                  // ldsP row stride

static __device__ __forceinline__ __bf16 tobf(float f) {
  unsigned u = __builtin_bit_cast(unsigned, f);
  unsigned r = u + 0x7FFFu + ((u >> 16) & 1u);            // RNE
  unsigned short h = (unsigned short)(r >> 16);
  return __builtin_bit_cast(__bf16, h);
}

// Pack two f32 -> two bf16; single v_cvt_pk_bf16_f32 when available.
static __device__ __forceinline__ v2bf pk2(float a, float b) {
#if __has_builtin(__builtin_amdgcn_cvt_pk_bf16_f32)
  return __builtin_amdgcn_cvt_pk_bf16_f32(a, b);
#else
  v2bf r; r[0] = tobf(a); r[1] = tobf(b); return r;
#endif
}

static __device__ __forceinline__ v8bf cvt8(float4 a, float4 b) {
  v4bf lo = __builtin_shufflevector(pk2(a.x, a.y), pk2(a.z, a.w), 0, 1, 2, 3);
  v4bf hi = __builtin_shufflevector(pk2(b.x, b.y), pk2(b.z, b.w), 0, 1, 2, 3);
  return __builtin_shufflevector(lo, hi, 0, 1, 2, 3, 4, 5, 6, 7);
}

// Build a 16-elem bf16 fragment from two 16-byte LDS loads (p and p+off2 elems).
static __device__ __forceinline__ v16bf ldfrag(const __bf16* p, int off2) {
  v8bf lo = *(const v8bf*)(p);
  v8bf hi = *(const v8bf*)(p + off2);
  return __builtin_shufflevector(lo, hi, 0, 1, 2, 3, 4, 5, 6, 7,
                                         8, 9, 10, 11, 12, 13, 14, 15);
}

// DPP16 cross-lane move (stays inside each 16-lane row on wave32).
template <int CTRL>
static __device__ __forceinline__ float dppf(float x) {
  int i = __builtin_bit_cast(int, x);
  int r = __builtin_amdgcn_update_dpp(i, i, CTRL, 0xF, 0xF, false);
  return __builtin_bit_cast(float, r);
}
// Full 16-lane reduction, result broadcast to all 16 lanes of the row:
// xor1 = quad_perm(1,0,3,2)=0xB1, xor2 = quad_perm(2,3,0,1)=0x4E,
// then row_ror:4 (0x124) and row_ror:8 (0x128) fold the quads.
static __device__ __forceinline__ float redmax16(float t) {
  t = fmaxf(t, dppf<0xB1>(t));
  t = fmaxf(t, dppf<0x4E>(t));
  t = fmaxf(t, dppf<0x124>(t));
  t = fmaxf(t, dppf<0x128>(t));
  return t;
}
static __device__ __forceinline__ float redadd16(float t) {
  t += dppf<0xB1>(t);
  t += dppf<0x4E>(t);
  t += dppf<0x124>(t);
  t += dppf<0x128>(t);
  return t;
}

__global__ __launch_bounds__(256, 1)
void dilated_flash_attn_kernel(const float* __restrict__ q,
                               const float* __restrict__ k,
                               const float* __restrict__ v,
                               float* __restrict__ out) {
  // Double-buffered K/V tiles + per-wave Q staging / P relayout.  ~83 KB total.
  __shared__ alignas(16) __bf16 ldsK[2][BC * KS];     // 2 x  8704 B
  __shared__ alignas(16) __bf16 ldsV[2][HD * VS];     // 2 x 10240 B ([d][key])
  __shared__ alignas(16) __bf16 ldsQ[8 * 16 * QS];    // 34816 B
  __shared__ alignas(16) __bf16 ldsP[8 * 16 * PS];    // 10240 B

  const int tid  = threadIdx.x;
  const int lane = tid & 31;
  const int wave = tid >> 5;
  const int hf   = lane >> 4;     // lane half (0/1)
  const int l15  = lane & 15;

  // ---- map blockIdx.y -> (head, token base, token stride, dilation) ----
  const int prob = blockIdx.y;
  int head, base, stride, zoff;
  bool dil;
  if (prob < 32) {                      // group 0: seg_len 2048, r = 1
    head = prob >> 2;
    int seg = prob & 3;
    base = seg * 2048; stride = 1; dil = false; zoff = 0;
  } else {                              // group 1: seg_len 4096, r = 2, gidx = 1
    int p = prob - 32;
    head = 8 + (p >> 1);
    int seg = p & 1;
    int off = (1 + seg) & 1;            // offsets = (1 + seg) % 2
    base = seg * 4096 + off; stride = 2; dil = true; zoff = 1 - 2 * off;
  }

  const int j0 = blockIdx.x * BR + wave * 16;   // first dilated q-row of this wave
  const long hoff = (long)head * HD;

  // ---- stage this wave's 16x128 Q rows as bf16 into its LDS region ----
  {
    int row = lane >> 1;                // 0..15 (2 lanes per row)
    int dstart = (lane & 1) * 64;       // each lane converts 64 d-values
    long gt = (long)base + (long)(j0 + row) * stride;
    const float* qp = q + gt * ROWSTRIDE + hoff + dstart;
    __bf16* dst = &ldsQ[(wave * 16 + row) * QS + dstart];
#pragma unroll
    for (int i = 0; i < 8; ++i) {
      float4 a = ((const float4*)qp)[2 * i];
      float4 b = ((const float4*)qp)[2 * i + 1];
      *(v8bf*)(dst + i * 8) = cvt8(a, b);
    }
  }

  // ---- KV staging maps & pipelined load/store helpers ----
  const int ktok = tid >> 3;            // K staging: token 0..31, 16 d per thread
  const int dj   = (tid & 7) * 16;
  const int vtok = (tid >> 4) * 2;      // V staging: 2 tokens, 8 d per thread
  const int vd   = (tid & 15) * 8;

  const long stepT = (long)BC * stride * ROWSTRIDE;   // floats per kv tile
  const float* kp0  = k + ((long)base + (long)ktok * stride) * ROWSTRIDE + hoff + dj;
  const float* vp0  = v + ((long)base + (long)vtok * stride) * ROWSTRIDE + hoff + vd;
  const float* vp1  = vp0 + (long)stride * ROWSTRIDE;

  float4 rK[4], rV[4];
  auto loadKV = [&](int kt) {
    const float* kp = kp0 + kt * stepT;
    rK[0] = ((const float4*)kp)[0]; rK[1] = ((const float4*)kp)[1];
    rK[2] = ((const float4*)kp)[2]; rK[3] = ((const float4*)kp)[3];
    const float* va = vp0 + kt * stepT;
    const float* vb = vp1 + kt * stepT;
    rV[0] = ((const float4*)va)[0]; rV[1] = ((const float4*)va)[1];
    rV[2] = ((const float4*)vb)[0]; rV[3] = ((const float4*)vb)[1];
    if (kt + 1 < NT) {                  // prefetch one more tile ahead
      __builtin_prefetch(kp + stepT, 0, 1);
      __builtin_prefetch(va + stepT, 0, 1);
      __builtin_prefetch(vb + stepT, 0, 1);
    }
  };
  auto storeKV = [&](int b) {
    __bf16* bk = &ldsK[b][ktok * KS + dj];
    *(v8bf*)(bk)     = cvt8(rK[0], rK[1]);
    *(v8bf*)(bk + 8) = cvt8(rK[2], rK[3]);
    __bf16* bv = &ldsV[b][vd * VS + vtok];
    *(v2bf*)(bv + 0 * VS) = pk2(rV[0].x, rV[2].x);
    *(v2bf*)(bv + 1 * VS) = pk2(rV[0].y, rV[2].y);
    *(v2bf*)(bv + 2 * VS) = pk2(rV[0].z, rV[2].z);
    *(v2bf*)(bv + 3 * VS) = pk2(rV[0].w, rV[2].w);
    *(v2bf*)(bv + 4 * VS) = pk2(rV[1].x, rV[3].x);
    *(v2bf*)(bv + 5 * VS) = pk2(rV[1].y, rV[3].y);
    *(v2bf*)(bv + 6 * VS) = pk2(rV[1].z, rV[3].z);
    *(v2bf*)(bv + 7 * VS) = pk2(rV[1].w, rV[3].w);
  };

  // prologue: tile 0 into buffer 0 (Q staging synced by the same barrier)
  loadKV(0);
  storeKV(0);
  __syncthreads();

  // Q A-fragments (16-bit A layout: base K = hf*8, second half +16).
  v16bf aq[4];
#pragma unroll
  for (int c = 0; c < 4; ++c)
    aq[c] = ldfrag(&ldsQ[(wave * 16 + l15) * QS + c * 32 + hf * 8], 16);

  v8f acc[8];
#pragma unroll
  for (int t = 0; t < 8; ++t) acc[t] = v8f{};
  float mrow[8], lrow[8];
#pragma unroll
  for (int g = 0; g < 8; ++g) { mrow[g] = -3.0e38f; lrow[g] = 0.0f; }

  const float SL2E = 0.08838834764831845f * 1.4426950408889634f;  // (1/sqrt(128))*log2(e)

  for (int kt = 0; kt < NT; ++kt) {
    const int cur = kt & 1;
    const bool more = (kt + 1 < NT);
    if (more) loadKV(kt + 1);           // issue next tile's global loads now

    const __bf16* Kc = ldsK[cur];
    const __bf16* Vc = ldsV[cur];

    // ---- S = Q * K^T : two 16x16 tiles (keys 0-15, 16-31), 4 k-chunks ----
    v8f S0 = v8f{}, S1 = v8f{};
#pragma unroll
    for (int c = 0; c < 4; ++c) {
      v16bf b0 = ldfrag(&Kc[(l15)      * KS + c * 32 + hf * 8], 16);
      v16bf b1 = ldfrag(&Kc[(16 + l15) * KS + c * 32 + hf * 8], 16);
      S0 = __builtin_amdgcn_wmma_f32_16x16x32_bf16(false, aq[c], false, b0,
                                                   (short)0, S0, false, false);
      S1 = __builtin_amdgcn_wmma_f32_16x16x32_bf16(false, aq[c], false, b1,
                                                   (short)0, S1, false, false);
    }

    // ---- online softmax: C layout row = g + hf*8, col = l15; DPP reductions ----
    v8f alphav;
#pragma unroll
    for (int g = 0; g < 8; ++g) {
      float t = redmax16(fmaxf(S0[g], S1[g]));
      float mn = fmaxf(mrow[g], t);
      float al = __builtin_amdgcn_exp2f((mrow[g] - mn) * SL2E);
      mrow[g] = mn;
      float p0 = __builtin_amdgcn_exp2f((S0[g] - mn) * SL2E);
      float p1 = __builtin_amdgcn_exp2f((S1[g] - mn) * SL2E);
      float rs = redadd16(p0 + p1);
      lrow[g] = lrow[g] * al + rs;
      alphav[g] = al;
      int row = g + hf * 8;
      ldsP[(wave * 16 + row) * PS + l15]      = tobf(p0);
      ldsP[(wave * 16 + row) * PS + 16 + l15] = tobf(p1);
    }
#pragma unroll
    for (int t = 0; t < 8; ++t) acc[t] *= alphav;

    // ---- O += P * V ----
    v16bf ap = ldfrag(&ldsP[(wave * 16 + l15) * PS + hf * 8], 16);
#pragma unroll
    for (int t = 0; t < 8; ++t) {
      v16bf bv = ldfrag(&Vc[(t * 16 + l15) * VS + hf * 8], 16);
      acc[t] = __builtin_amdgcn_wmma_f32_16x16x32_bf16(false, ap, false, bv,
                                                       (short)0, acc[t], false, false);
    }

    // convert + stash next tile into the other buffer (loads have had the whole
    // compute section to land), then one barrier for both publish & protect.
    if (more) storeKV(cur ^ 1);
    __syncthreads();
  }

  // ---- epilogue: O /= l, scatter to (b, n, h, d); zero partner parity if dilated ----
#pragma unroll
  for (int g = 0; g < 8; ++g) {
    int row = g + hf * 8;
    float rinv = __builtin_amdgcn_rcpf(lrow[g]);
    long gt = (long)base + (long)(j0 + row) * stride;
    float* op = out + gt * ROWSTRIDE + hoff;
#pragma unroll
    for (int t = 0; t < 8; ++t) op[t * 16 + l15] = acc[t][g] * rinv;
    if (dil) {
      float* zp = out + (gt + zoff) * ROWSTRIDE + hoff;
#pragma unroll
      for (int t = 0; t < 8; ++t) zp[t * 16 + l15] = 0.0f;
    }
  }
}

extern "C" void kernel_launch(void* const* d_in, const int* in_sizes, int n_in,
                              void* d_out, int out_size, void* d_ws, size_t ws_size,
                              hipStream_t stream) {
  (void)in_sizes; (void)n_in; (void)out_size; (void)d_ws; (void)ws_size;
  const float* q = (const float*)d_in[0];
  const float* k = (const float*)d_in[1];
  const float* v = (const float*)d_in[2];
  float* out = (float*)d_out;
  dim3 grid(SEQ / BR, 48, 1);
  dim3 block(256, 1, 1);
  hipLaunchKernelGGL(dilated_flash_attn_kernel, grid, block, 0, stream, q, k, v, out);
}
